// gat_66632122630727
// MI455X (gfx1250) — matrix operand order repaired and checked by reading, not compiled, not run on hardware
//
#include <hip/hip_runtime.h>

#define NN 100000
#define NE 1600000
#define DD 256
#define HH 4
#define CC 64
#define LL 3

typedef _Float16 half8 __attribute__((ext_vector_type(8)));
typedef _Float16 v16h  __attribute__((ext_vector_type(16)));
typedef float    v8f   __attribute__((ext_vector_type(8)));

__device__ __forceinline__ int f2ord(float f) {
    int i = __float_as_int(f);
    return i >= 0 ? i : (i ^ 0x7fffffff);
}
__device__ __forceinline__ float ord2f(int i) {
    return __int_as_float(i >= 0 ? i : (i ^ 0x7fffffff));
}

// ---------------------------------------------------------------------------
// WMMA GEMM: C[M,256] = op(A[M,256] @ W[256,256] + bias)
// Block: 256 threads (8 waves). Tile: 128 rows x 64 cols, K staged in f16 LDS.
// Double-buffered LDS software pipeline, fully unrolled (8 K-steps, 32 wmma).
// Each wave: 16 rows x 64 cols = 4 accumulator tiles of v_wmma_f32_16x16x32_f16.
// ---------------------------------------------------------------------------
__global__ __launch_bounds__(256) void gemm256_wmma(
    const float* __restrict__ A, const float* __restrict__ W,
    const float* __restrict__ bias, float* __restrict__ C,
    int M, int do_relu)
{
    __shared__ __align__(16) _Float16 sA[2][128 * 40];  // 128 rows x 32 k (pad->40)
    __shared__ __align__(16) _Float16 sB[2][64 * 40];   // 64 cols x 32 k transposed (pad->40)

    const int tid  = threadIdx.x;
    const int lane = tid & 31;
    const int wv   = tid >> 5;
    const int rm   = blockIdx.x * 128;
    const int nb   = blockIdx.y * 64;

    // staging coordinates (branchless row clamp: rows >= M read row M-1, never stored)
    const int aRow = tid >> 1;                 // with i-offset: 2 rows per i-step? (see below)
    v8f acc[4];
    acc[0] = v8f{}; acc[1] = v8f{}; acc[2] = v8f{}; acc[3] = v8f{};

    float4 ra[4], rb[2];

    // A: 128 rows x 8 float4; thread t, iter i -> lin = i*256+t; row=lin>>3, c4=lin&7
    // B: 32 k-rows x 16 float4; iter i -> lin = i*256+t; kr=lin>>4, c4=lin&15
    auto load_regs = [&](int kb) {
        #pragma unroll
        for (int i = 0; i < 4; i++) {
            int lin = i * 256 + tid;
            int row = lin >> 3, c4 = lin & 7;
            int gr = rm + row;
            gr = gr < M ? gr : (M - 1);
            ra[i] = *(const float4*)(A + (size_t)gr * 256 + kb + c4 * 4);
        }
        #pragma unroll
        for (int i = 0; i < 2; i++) {
            int lin = i * 256 + tid;
            int kr = lin >> 4, c4 = lin & 15;
            rb[i] = *(const float4*)(W + (size_t)(kb + kr) * 256 + nb + c4 * 4);
        }
    };
    (void)aRow;

    load_regs(0);

    #pragma unroll
    for (int step = 0; step < 8; step++) {
        const int buf = step & 1;
        // ---- convert staged registers -> f16 LDS ----
        #pragma unroll
        for (int i = 0; i < 4; i++) {
            int lin = i * 256 + tid;
            int row = lin >> 3, c4 = lin & 7;
            _Float16* d = &sA[buf][row * 40 + c4 * 4];
            d[0] = (_Float16)ra[i].x; d[1] = (_Float16)ra[i].y;
            d[2] = (_Float16)ra[i].z; d[3] = (_Float16)ra[i].w;
        }
        #pragma unroll
        for (int i = 0; i < 2; i++) {
            int lin = i * 256 + tid;
            int kr = lin >> 4, c4 = lin & 15;
            sB[buf][(c4 * 4 + 0) * 40 + kr] = (_Float16)rb[i].x;
            sB[buf][(c4 * 4 + 1) * 40 + kr] = (_Float16)rb[i].y;
            sB[buf][(c4 * 4 + 2) * 40 + kr] = (_Float16)rb[i].z;
            sB[buf][(c4 * 4 + 3) * 40 + kr] = (_Float16)rb[i].w;
        }
        // ---- issue next tile's global loads before the barrier (overlap with wmma) ----
        if (step < 7) load_regs((step + 1) * 32);
        __syncthreads();

        // ---- fragments: A 16x32 (lane<16: row=lane, K{0..7,16..23}; lane>=16: K{8..15,24..31})
        int arow  = (wv << 4) + (lane & 15);
        int khalf = (lane >> 4) * 8;
        half8 alo = *(const half8*)&sA[buf][arow * 40 + khalf];
        half8 ahi = *(const half8*)&sA[buf][arow * 40 + khalf + 16];
        v16h a = __builtin_shufflevector(alo, ahi, 0,1,2,3,4,5,6,7,8,9,10,11,12,13,14,15);

        // B 32x16 (lane<16: col=lane, K{0..7,16..23}; lane>=16: col=lane-16, K{8..15,24..31})
        int kseg = (lane >> 4) * 8;
        v16h b[4];
        #pragma unroll
        for (int t = 0; t < 4; t++) {
            int bcol = t * 16 + (lane & 15);
            half8 blo = *(const half8*)&sB[buf][bcol * 40 + kseg];
            half8 bhi = *(const half8*)&sB[buf][bcol * 40 + kseg + 16];
            b[t] = __builtin_shufflevector(blo, bhi, 0,1,2,3,4,5,6,7,8,9,10,11,12,13,14,15);
        }
        #pragma unroll
        for (int t = 0; t < 4; t++) {
            acc[t] = __builtin_amdgcn_wmma_f32_16x16x32_f16(
                false, a, false, b[t], (short)0, acc[t], false, false);
        }
        __syncthreads();
    }

    // C/D layout: VGPR r, lane L -> row = r + 8*(L/16), col = L%16
    #pragma unroll
    for (int t = 0; t < 4; t++) {
        #pragma unroll
        for (int r = 0; r < 8; r++) {
            int row = rm + (wv << 4) + r + ((lane >> 4) << 3);
            int col = nb + t * 16 + (lane & 15);
            if (row < M) {
                float v = acc[t][r];
                if (bias) v += bias[col];
                if (do_relu) v = fmaxf(v, 0.f);
                C[(size_t)row * 256 + col] = v;
            }
        }
    }
}

// ---------------------------------------------------------------------------
// Per-node attention logits: asrc[n,h] = <xt[n,h*64:...], att_src[h]>, same for dst.
// 64 nodes per block staged through LDS.
// ---------------------------------------------------------------------------
__global__ __launch_bounds__(256) void node_alpha_kernel(
    const float* __restrict__ xt, const float* __restrict__ a_s,
    const float* __restrict__ a_d, float* __restrict__ asrc,
    float* __restrict__ adst, int n)
{
    __shared__ float sx[64 * 256];  // 64 KB
    int base = blockIdx.x * 64;
    #pragma unroll
    for (int i = 0; i < 16; i++) {
        int lin = i * 256 + threadIdx.x;   // 4096 float4
        int row = lin >> 6;                // 64 float4 per row
        int c4  = lin & 63;
        int gr  = base + row;
        gr = gr < n ? gr : (n - 1);        // branchless clamp; dup rows are harmless
        *(float4*)&sx[row * 256 + c4 * 4] =
            *(const float4*)(xt + (size_t)gr * 256 + c4 * 4);
    }
    __syncthreads();
    int node = threadIdx.x >> 2;
    int h    = threadIdx.x & 3;
    if (base + node < n) {
        const float* xp = &sx[node * 256 + h * 64];
        const float* ps = a_s + h * 64;
        const float* pd = a_d + h * 64;
        float s = 0.f, d = 0.f;
        #pragma unroll
        for (int c = 0; c < 64; c++) {
            float xv = xp[c];
            s = fmaf(xv, ps[c], s);
            d = fmaf(xv, pd[c], d);
        }
        asrc[(size_t)(base + node) * 4 + h] = s;
        adst[(size_t)(base + node) * 4 + h] = d;
    }
}

// v[l,k,h] = sum_c We[l,k,h*64+c] * att_edge[l,h,c]  (folds the whole et tensor away)
__global__ void ve_kernel(const float* __restrict__ We,
                          const float* __restrict__ att_edge,
                          float* __restrict__ ve)
{
    int id = threadIdx.x;
    if (id < LL * 3 * HH) {
        int l = id / 12, k = (id % 12) / 4, h = id % 4;
        const float* w = We + ((size_t)l * 3 + k) * 256 + h * 64;
        const float* a = att_edge + (size_t)l * 256 + h * 64;
        float s = 0.f;
        #pragma unroll
        for (int c = 0; c < 64; c++) s = fmaf(w[c], a[c], s);
        ve[id] = s;
    }
}

__global__ void init_layer_kernel(float* __restrict__ agg, int* __restrict__ amax,
                                  float* __restrict__ denom, int ntot, int nsm)
{
    int i = blockIdx.x * 256 + threadIdx.x;
    if (i < ntot) agg[i] = 0.f;
    if (i < nsm) { amax[i] = f2ord(-3.0e38f); denom[i] = 0.f; }
}

// Pass A: alpha = leaky_relu(asrc[src] + adst[dst] + ea@v), segment max via ordered-int atomicMax
__global__ __launch_bounds__(256) void edge_alpha_kernel(
    const int* __restrict__ src, const int* __restrict__ dst,
    const float* __restrict__ ea, const float* __restrict__ asrc,
    const float* __restrict__ adst, const float* __restrict__ ve,
    float* __restrict__ alpha, int* __restrict__ amax, int E)
{
    int e = blockIdx.x * 256 + threadIdx.x;
    if (e >= E) return;
    int s = src[e], d = dst[e];
    float e0 = ea[(size_t)e * 3 + 0];
    float e1 = ea[(size_t)e * 3 + 1];
    float e2 = ea[(size_t)e * 3 + 2];
    float4 s4 = *(const float4*)(asrc + (size_t)s * 4);
    float4 d4 = *(const float4*)(adst + (size_t)d * 4);
    float base[4] = { s4.x + d4.x, s4.y + d4.y, s4.z + d4.z, s4.w + d4.w };
    #pragma unroll
    for (int h = 0; h < 4; h++) {
        float a = base[h] + e0 * ve[h] + e1 * ve[4 + h] + e2 * ve[8 + h];
        a = a > 0.f ? a : 0.2f * a;   // leaky_relu(0.2)
        alpha[(size_t)e * 4 + h] = a;
        atomicMax(&amax[(size_t)d * 4 + h], f2ord(a));
    }
}

// Pass B: ex = exp(alpha - amax[dst]); denom[dst] += ex
__global__ __launch_bounds__(256) void edge_exp_kernel(
    const int* __restrict__ dst, float* __restrict__ alpha,
    const int* __restrict__ amax, float* __restrict__ denom, int E)
{
    int e = blockIdx.x * 256 + threadIdx.x;
    if (e >= E) return;
    int d = dst[e];
    float4 a = *(const float4*)(alpha + (size_t)e * 4);
    float m0 = ord2f(amax[(size_t)d * 4 + 0]);
    float m1 = ord2f(amax[(size_t)d * 4 + 1]);
    float m2 = ord2f(amax[(size_t)d * 4 + 2]);
    float m3 = ord2f(amax[(size_t)d * 4 + 3]);
    float x0 = __expf(a.x - m0), x1 = __expf(a.y - m1);
    float x2 = __expf(a.z - m2), x3 = __expf(a.w - m3);
    *(float4*)(alpha + (size_t)e * 4) = make_float4(x0, x1, x2, x3);
    atomicAdd(&denom[(size_t)d * 4 + 0], x0);
    atomicAdd(&denom[(size_t)d * 4 + 1], x1);
    atomicAdd(&denom[(size_t)d * 4 + 2], x2);
    atomicAdd(&denom[(size_t)d * 4 + 3], x3);
}

// Pass C: agg[dst] += xt[src] * coef. 64 threads/edge (2 waves): e is wave-uniform,
// so force src/dst through readfirstlane -> scalar loads via the constant cache.
__global__ __launch_bounds__(256) void edge_scatter_kernel(
    const int* __restrict__ src, const int* __restrict__ dst,
    const float* __restrict__ ex, const float* __restrict__ denom,
    const float* __restrict__ xt, float* __restrict__ agg, int E)
{
    int gid = blockIdx.x * 256 + threadIdx.x;
    if (gid >= E * 64) return;
    int e = __builtin_amdgcn_readfirstlane(gid >> 6);  // wave-uniform edge id
    int t = gid & 63;       // 64 threads cover 256 cols (4 each)
    int h = t >> 4;
    int s = __builtin_amdgcn_readfirstlane(src[e]);
    int d = __builtin_amdgcn_readfirstlane(dst[e]);
    float coef = ex[(size_t)e * 4 + h] / denom[(size_t)d * 4 + h];
    float4 v = *(const float4*)(xt + (size_t)s * 256 + t * 4);
    float* o = agg + (size_t)d * 256 + t * 4;
    atomicAdd(o + 0, v.x * coef);
    atomicAdd(o + 1, v.y * coef);
    atomicAdd(o + 2, v.z * coef);
    atomicAdd(o + 3, v.w * coef);
}

// x = relu(agg + bias) + x   (residual)
__global__ void epilogue_kernel(const float* __restrict__ agg,
                                const float* __restrict__ bias,
                                float* __restrict__ x, int ntot)
{
    int i = blockIdx.x * 256 + threadIdx.x;
    if (i < ntot) {
        float v = agg[i] + bias[i & 255];
        x[i] = fmaxf(v, 0.f) + x[i];
    }
}

extern "C" void kernel_launch(void* const* d_in, const int* in_sizes, int n_in,
                              void* d_out, int out_size, void* d_ws, size_t ws_size,
                              hipStream_t stream)
{
    const float* x        = (const float*)d_in[0];
    const float* edge_attr= (const float*)d_in[1];
    const float* W        = (const float*)d_in[2];
    const float* att_src  = (const float*)d_in[3];
    const float* att_dst  = (const float*)d_in[4];
    const float* We       = (const float*)d_in[5];
    const float* att_edge = (const float*)d_in[6];
    const float* bias     = (const float*)d_in[7];
    const float* W1       = (const float*)d_in[8];
    const float* b1       = (const float*)d_in[9];
    const float* W2       = (const float*)d_in[10];
    const float* b2       = (const float*)d_in[11];
    const int*   ei       = (const int*)d_in[12];
    const int* src = ei;
    const int* dst = ei + NE;
    float* out = (float*)d_out;

    // workspace partition (all 256B aligned)
    char* ws = (char*)d_ws;
    size_t off = 0;
    auto alloc = [&](size_t bytes) -> char* {
        char* p = ws + off;
        off += (bytes + 255) & ~(size_t)255;
        return p;
    };
    float* x_cur   = (float*)alloc((size_t)NN * 256 * 4);
    float* xt      = (float*)alloc((size_t)NN * 256 * 4);
    float* agg     = (float*)alloc((size_t)NN * 256 * 4);
    float* asrc    = (float*)alloc((size_t)NN * 4 * 4);
    float* adst    = (float*)alloc((size_t)NN * 4 * 4);
    int*   amax    = (int*)  alloc((size_t)NN * 4 * 4);
    float* denom   = (float*)alloc((size_t)NN * 4 * 4);
    float* alphaws = (float*)alloc((size_t)NE * 4 * 4);
    float* ve      = (float*)alloc(LL * 3 * HH * 4);

    hipMemcpyAsync(x_cur, x, (size_t)NN * 256 * 4, hipMemcpyDeviceToDevice, stream);
    ve_kernel<<<1, 64, 0, stream>>>(We, att_edge, ve);

    dim3 ggemm((NN + 127) / 128, 4);
    int nblkN256 = (NN * 256 + 255) / 256;
    int nblkE    = (NE + 255) / 256;
    int nblkE64  = (NE * 64 + 255) / 256;

    for (int l = 0; l < LL; l++) {
        gemm256_wmma<<<ggemm, 256, 0, stream>>>(
            x_cur, W + (size_t)l * 256 * 256, nullptr, xt, NN, 0);
        node_alpha_kernel<<<(NN + 63) / 64, 256, 0, stream>>>(
            xt, att_src + (size_t)l * 256, att_dst + (size_t)l * 256, asrc, adst, NN);
        init_layer_kernel<<<nblkN256, 256, 0, stream>>>(agg, amax, denom, NN * 256, NN * 4);
        edge_alpha_kernel<<<nblkE, 256, 0, stream>>>(
            src, dst, edge_attr, asrc, adst, ve + l * 12, alphaws, amax, NE);
        edge_exp_kernel<<<nblkE, 256, 0, stream>>>(dst, alphaws, amax, denom, NE);
        edge_scatter_kernel<<<nblkE64, 256, 0, stream>>>(
            src, dst, alphaws, denom, xt, agg, NE);
        epilogue_kernel<<<nblkN256, 256, 0, stream>>>(
            agg, bias + (size_t)l * 256, x_cur, NN * 256);
    }

    // MLP: out = relu(x @ W1 + b1) @ W2 + b2
    gemm256_wmma<<<ggemm, 256, 0, stream>>>(x_cur, W1, b1, xt, NN, 1);
    gemm256_wmma<<<ggemm, 256, 0, stream>>>(xt, W2, b2, out, NN, 0);
}
